// SUN_72069551226903
// MI455X (gfx1250) — compile-verified, matrix operand order checked by reading, not run on hardware
//
#include <hip/hip_runtime.h>

typedef __attribute__((ext_vector_type(16))) __bf16 v16bf;
typedef __attribute__((ext_vector_type(8)))  float  v8f;
typedef __attribute__((ext_vector_type(4)))  int    v4i;

#define NH    4096
#define NIN   784
#define NOUT  10
#define NEDGE 200000
#define BATCH 4096

#define TM 128
#define TN 128
#define TK 32
#define LSTR 40   // bf16 elements per LDS row: 32 + 8 pad -> 80B rows, 16B aligned, bank-conflict-free

#if __has_builtin(__builtin_amdgcn_global_load_async_to_lds_b128) && \
    __has_builtin(__builtin_amdgcn_s_wait_asynccnt)
#define USE_ASYNC_LDS 1
#else
#define USE_ASYNC_LDS 0
#endif

// ---------------- W build: scatter-add (coalesce duplicates) in fp32 ----------------
__global__ void scatter_edges(const int* __restrict__ idx,
                              const float* __restrict__ val,
                              float* __restrict__ W) {
    int i = blockIdx.x * blockDim.x + threadIdx.x;
    if (i < NEDGE) {
        int r = idx[i];
        int c = idx[i + NEDGE];
        atomicAdd(&W[(size_t)r * NH + c], val[i]);
    }
}

// ---------------- transpose + convert: Wt[n][k] = bf16(W[k][n]) ----------------
__global__ void transpose_cvt(const float* __restrict__ W, __bf16* __restrict__ Wt) {
    __shared__ float t[32][33];
    int k0 = blockIdx.x * 32, n0 = blockIdx.y * 32;
    int tx = threadIdx.x, ty = threadIdx.y;   // 32 x 8
#pragma unroll
    for (int i = 0; i < 32; i += 8)
        t[ty + i][tx] = W[(size_t)(k0 + ty + i) * NH + (n0 + tx)];
    __syncthreads();
#pragma unroll
    for (int i = 0; i < 32; i += 8)
        Wt[(size_t)(n0 + ty + i) * NH + (k0 + tx)] = (__bf16)t[tx][ty + i];
}

// ---------------- hs0: inputs injected into first NIN neurons ----------------
__global__ void init_hs(const float* __restrict__ inp, __bf16* __restrict__ hs) {
    size_t i = (size_t)blockIdx.x * blockDim.x + threadIdx.x;
    int col = (int)(i & (NH - 1));
    size_t row = i >> 12;
    float v = (col < NIN) ? inp[row * NIN + col] : 0.0f;
    hs[i] = (__bf16)v;
}

// ---------------- main WMMA GEMM step: C = act(A @ Wt^T) ----------------
// A: [BATCH x NH] bf16 row-major; Bt: Wt [NH x NH] bf16 (n-major, i.e. W transposed)
__global__ __launch_bounds__(256, 2)
void gemm_step(const __bf16* __restrict__ A, const __bf16* __restrict__ Bt,
               __bf16* __restrict__ C) {
    __shared__ __align__(16) __bf16 lA[2][TM * LSTR];
    __shared__ __align__(16) __bf16 lB[2][TN * LSTR];

    const int tid  = threadIdx.x;
    const int lane = tid & 31;
    const int wv   = tid >> 5;      // 0..7
    const int wm   = wv >> 1;       // 0..3 -> M offset wm*32
    const int wn   = wv & 1;        // 0..1 -> N offset wn*64
    const int half = lane >> 4;     // lane 0-15 / 16-31
    const int l15  = lane & 15;

    const int mblk = blockIdx.x * TM;
    const int nblk = blockIdx.y * TN;

    // global->LDS staging: 512 chunks of 16B per tile; this thread owns chunk tid and tid+256
    const int r0 = tid >> 2,         s0 = tid & 3;
    const int r1 = (tid + 256) >> 2, s1 = (tid + 256) & 3;
    const __bf16* Ab0 = A  + (size_t)(mblk + r0) * NH + s0 * 8;
    const __bf16* Ab1 = A  + (size_t)(mblk + r1) * NH + s1 * 8;
    const __bf16* Bb0 = Bt + (size_t)(nblk + r0) * NH + s0 * 8;
    const __bf16* Bb1 = Bt + (size_t)(nblk + r1) * NH + s1 * 8;
    const int la0 = r0 * LSTR + s0 * 8;
    const int la1 = r1 * LSTR + s1 * 8;

    // per-lane fragment LDS offsets (ISA 16-bit A/B layout: K chunks half*8 and 16+half*8)
    int aoff[2], boff[4];
#pragma unroll
    for (int i = 0; i < 2; ++i) aoff[i] = (wm * 32 + i * 16 + l15) * LSTR + half * 8;
#pragma unroll
    for (int j = 0; j < 4; ++j) boff[j] = (wn * 64 + j * 16 + l15) * LSTR + half * 8;

    v8f acc[2][4] = {};
    union Frag { uint4 q[2]; v16bf v; };
    const int KT = NH / TK;   // 128

#if USE_ASYNC_LDS
    // ---- async global->LDS copies (ASYNCcnt path), no VGPR staging ----
    // builtin signature (from toolchain diagnostic): (v4i as1*, v4i as3*, imm offset, imm cpol)
    #define ASYNC16(gp, lp)                                                          \
        __builtin_amdgcn_global_load_async_to_lds_b128(                              \
            (__attribute__((address_space(1))) v4i*)(gp),                            \
            (__attribute__((address_space(3))) v4i*)(lp), 0, 0)

    // prologue: stage 0 into lds buffer 0
    ASYNC16(Ab0, &lA[0][la0]);  ASYNC16(Ab1, &lA[0][la1]);
    ASYNC16(Bb0, &lB[0][la0]);  ASYNC16(Bb1, &lB[0][la1]);
    __builtin_amdgcn_s_wait_asynccnt(0);
    __syncthreads();

    for (int kt = 0; kt < KT; ++kt) {
        const int buf = kt & 1;
        const bool more = (kt + 1) < KT;
        if (more) {
            const int nb = buf ^ 1;
            const int ko = (kt + 1) * TK;
            ASYNC16(Ab0 + ko, &lA[nb][la0]);  ASYNC16(Ab1 + ko, &lA[nb][la1]);
            ASYNC16(Bb0 + ko, &lB[nb][la0]);  ASYNC16(Bb1 + ko, &lB[nb][la1]);
        }

        Frag fa[2], fb[4];
#pragma unroll
        for (int i = 0; i < 2; ++i) {
            fa[i].q[0] = *(const uint4*)&lA[buf][aoff[i]];
            fa[i].q[1] = *(const uint4*)&lA[buf][aoff[i] + 16];
        }
#pragma unroll
        for (int j = 0; j < 4; ++j) {
            fb[j].q[0] = *(const uint4*)&lB[buf][boff[j]];
            fb[j].q[1] = *(const uint4*)&lB[buf][boff[j] + 16];
        }

#pragma unroll
        for (int i = 0; i < 2; ++i)
#pragma unroll
            for (int j = 0; j < 4; ++j)
                acc[i][j] = __builtin_amdgcn_wmma_f32_16x16x32_bf16(
                    false, fa[i].v, false, fb[j].v, (short)0, acc[i][j], false, false);

        if (more) {
            __builtin_amdgcn_s_wait_asynccnt(0);
            __syncthreads();
        }
    }
    #undef ASYNC16
#else
    // ---- fallback: register-staged synchronous copies ----
    uint4 va0 = *(const uint4*)(Ab0);
    uint4 va1 = *(const uint4*)(Ab1);
    uint4 vb0 = *(const uint4*)(Bb0);
    uint4 vb1 = *(const uint4*)(Bb1);
    *(uint4*)&lA[0][la0] = va0;  *(uint4*)&lA[0][la1] = va1;
    *(uint4*)&lB[0][la0] = vb0;  *(uint4*)&lB[0][la1] = vb1;
    __syncthreads();

    for (int kt = 0; kt < KT; ++kt) {
        const int buf = kt & 1;
        const bool more = (kt + 1) < KT;
        if (more) {
            const int ko = (kt + 1) * TK;
            va0 = *(const uint4*)(Ab0 + ko);
            va1 = *(const uint4*)(Ab1 + ko);
            vb0 = *(const uint4*)(Bb0 + ko);
            vb1 = *(const uint4*)(Bb1 + ko);
        }

        Frag fa[2], fb[4];
#pragma unroll
        for (int i = 0; i < 2; ++i) {
            fa[i].q[0] = *(const uint4*)&lA[buf][aoff[i]];
            fa[i].q[1] = *(const uint4*)&lA[buf][aoff[i] + 16];
        }
#pragma unroll
        for (int j = 0; j < 4; ++j) {
            fb[j].q[0] = *(const uint4*)&lB[buf][boff[j]];
            fb[j].q[1] = *(const uint4*)&lB[buf][boff[j] + 16];
        }

#pragma unroll
        for (int i = 0; i < 2; ++i)
#pragma unroll
            for (int j = 0; j < 4; ++j)
                acc[i][j] = __builtin_amdgcn_wmma_f32_16x16x32_bf16(
                    false, fa[i].v, false, fb[j].v, (short)0, acc[i][j], false, false);

        if (more) {
            const int nb = buf ^ 1;
            *(uint4*)&lA[nb][la0] = va0;  *(uint4*)&lA[nb][la1] = va1;
            *(uint4*)&lB[nb][la0] = vb0;  *(uint4*)&lB[nb][la1] = vb1;
            __syncthreads();
        }
    }
#endif

    // epilogue: mask/relu split then bf16 store for the next step
#pragma unroll
    for (int i = 0; i < 2; ++i) {
        const int row0 = mblk + wm * 32 + i * 16 + half * 8;
#pragma unroll
        for (int j = 0; j < 4; ++j) {
            const int col = nblk + wn * 64 + j * 16 + l15;
            const bool raw = (col >= NH - NOUT);   // output neurons keep pre-activation
#pragma unroll
            for (int r = 0; r < 8; ++r) {
                float v = acc[i][j][r];
                if (!raw) v = fmaxf(v, 0.0f);
                C[(size_t)(row0 + r) * NH + col] = (__bf16)v;
            }
        }
    }
}

// ---------------- step 5 restricted to the 10 output columns + sigmoid ----------------
__global__ __launch_bounds__(256)
void final_step(const __bf16* __restrict__ hs, const __bf16* __restrict__ Wt,
                float* __restrict__ out) {
    __shared__ float red[NOUT][256];
    const int b = blockIdx.x;
    const int t = threadIdx.x;
    const __bf16* hrow = hs + (size_t)b * NH;
    float acc[NOUT] = {};
    for (int k = t; k < NH; k += 256) {
        float h = (float)hrow[k];
#pragma unroll
        for (int j = 0; j < NOUT; ++j)
            acc[j] += h * (float)Wt[(size_t)(NH - NOUT + j) * NH + k];
    }
#pragma unroll
    for (int j = 0; j < NOUT; ++j) red[j][t] = acc[j];
    __syncthreads();
    for (int s = 128; s > 0; s >>= 1) {
        if (t < s) {
#pragma unroll
            for (int j = 0; j < NOUT; ++j) red[j][t] += red[j][t + s];
        }
        __syncthreads();
    }
    if (t < NOUT)
        out[(size_t)b * NOUT + t] = 1.0f / (1.0f + __expf(-red[t][0]));
}

extern "C" void kernel_launch(void* const* d_in, const int* in_sizes, int n_in,
                              void* d_out, int out_size, void* d_ws, size_t ws_size,
                              hipStream_t stream) {
    (void)in_sizes; (void)n_in; (void)out_size; (void)ws_size;
    const float* inp = (const float*)d_in[0];
    const float* ev  = (const float*)d_in[1];
    const int*   ei  = (const int*)d_in[2];
    float* out = (float*)d_out;

    char* ws = (char*)d_ws;
    const size_t offWt = (size_t)NH * NH * 4;              //  64 MiB: W fp32 scratch before this
    const size_t offA  = offWt + (size_t)NH * NH * 2;      //  96 MiB
    const size_t offB  = offA  + (size_t)NH * NH * 2;      // 128 MiB (total 160 MiB used)
    float*  Wf  = (float*)ws;
    __bf16* Wt  = (__bf16*)(ws + offWt);
    __bf16* hsA = (__bf16*)(ws + offA);
    __bf16* hsB = (__bf16*)(ws + offB);

    (void)hipMemsetAsync(Wf, 0, (size_t)NH * NH * sizeof(float), stream);
    scatter_edges<<<(NEDGE + 255) / 256, 256, 0, stream>>>(ei, ev, Wf);
    transpose_cvt<<<dim3(NH / 32, NH / 32), dim3(32, 8), 0, stream>>>(Wf, Wt);
    init_hs<<<(size_t)BATCH * NH / 256, 256, 0, stream>>>(inp, hsA);

    dim3 gg(BATCH / TM, NH / TN);   // 32 x 32
    gemm_step<<<gg, 256, 0, stream>>>(hsA, Wt, hsB);   // step 1
    gemm_step<<<gg, 256, 0, stream>>>(hsB, Wt, hsA);   // step 2
    gemm_step<<<gg, 256, 0, stream>>>(hsA, Wt, hsB);   // step 3
    gemm_step<<<gg, 256, 0, stream>>>(hsB, Wt, hsA);   // step 4
    final_step<<<BATCH, 256, 0, stream>>>(hsA, Wt, out); // step 5 (only 10 cols) + sigmoid
}
